// SelfAttention_79199196938806
// MI455X (gfx1250) — compile-verified, matrix operand order checked by reading, not run on hardware
//
#include <hip/hip_runtime.h>

#define B_ 4
#define T_ 2048
#define C_ 1024
#define NH 16
#define HD 64
#define M_ (B_ * T_)

typedef __bf16 bf16;
typedef __attribute__((ext_vector_type(16))) __bf16 v16bf;
typedef __attribute__((ext_vector_type(8)))  float  v8f;

union Frag {
    v16bf v;
    uint4 u[2];
};

__device__ __forceinline__ v8f wmma_bf16(const Frag& a, const Frag& b, v8f c) {
    return __builtin_amdgcn_wmma_f32_16x16x32_bf16(false, a.v, false, b.v,
                                                   (short)0, c, false, false);
}

__device__ __forceinline__ v8f v8f_zero() {
    v8f z = {0.f, 0.f, 0.f, 0.f, 0.f, 0.f, 0.f, 0.f};
    return z;
}

// gfx1250 async copy: global memory -> LDS (16 bytes per lane), tracked by ASYNCcnt.
// LDS address = low 32 bits of generic pointer (ISA: LDS aperture truncates addr[31:0]).
__device__ __forceinline__ void async_copy16(void* lds_dst, const void* gsrc) {
    unsigned la = (unsigned)(unsigned long long)lds_dst;
    asm volatile("global_load_async_to_lds_b128 %0, %1, off"
                 :: "v"(la), "v"(gsrc) : "memory");
}

__device__ __forceinline__ void wait_async0() {
#if __has_builtin(__builtin_amdgcn_s_wait_asynccnt)
    __builtin_amdgcn_s_wait_asynccnt(0);
#else
    asm volatile("s_wait_asynccnt 0x0" ::: "memory");
#endif
}

// ---------------- prep kernels ----------------

__global__ void cast_f32_bf16(const float* __restrict__ src, bf16* __restrict__ dst, int n) {
    int i = blockIdx.x * blockDim.x + threadIdx.x;
    if (i < n) dst[i] = (bf16)src[i];
}

// src[R][Cc] row-major -> dst[Cc][R] (bf16), i.e. dst[n][k] = src[k][n]
__global__ void transpose_f32_bf16(const float* __restrict__ src, bf16* __restrict__ dst,
                                   int R, int Cc) {
    int i = blockIdx.x * blockDim.x + threadIdx.x;
    if (i < R * Cc) {
        int r = i / Cc, c = i - r * Cc;
        dst[(size_t)c * R + r] = (bf16)src[i];
    }
}

// ---------------- QKV GEMM: [M_,C_] x [C_,3C_] + bias, scatter to Q/K/Vt ----------------
// 128x128 block tile, 256 threads / 8 waves; wave computes 32x64 (8 WMMAs / K-step).
// Double-buffered async global->LDS pipeline: one barrier per K-step, copy of
// step k+1 overlaps WMMAs of step k.

__global__ __launch_bounds__(256) void qkv_gemm(const bf16* __restrict__ X,
                                                const bf16* __restrict__ Wt,
                                                const float* __restrict__ bias,
                                                bf16* __restrict__ Q,
                                                bf16* __restrict__ Kx,
                                                bf16* __restrict__ Vt) {
    __shared__ bf16 As[2][128][40];   // 128 rows x 32 K (+8 pad), 16B-aligned rows
    __shared__ bf16 Bs[2][128][40];
    const int tid  = threadIdx.x;
    const int wave = tid >> 5;
    const int lane = tid & 31;
    const int lr   = lane & 15;
    const int kof  = (lane >> 4) * 8;
    const int wm   = (wave >> 1) * 32;     // wave row offset in block: 0/32/64/96
    const int wn   = (wave & 1) * 64;      // wave col offset in block: 0/64
    const int m0   = blockIdx.x * 128;
    const int n0   = blockIdx.y * 128;

    v8f acc[2][4];
#pragma unroll
    for (int i = 0; i < 2; ++i)
#pragma unroll
        for (int j = 0; j < 4; ++j) acc[i][j] = v8f_zero();

    auto issue = [&](int k0, int bufi) {
#pragma unroll
        for (int i = 0; i < 2; ++i) {
            int c   = tid + 256 * i;       // 512 16B chunks per array
            int row = c >> 2;
            int kc  = (c & 3) * 8;
            async_copy16(&As[bufi][row][kc], &X[(size_t)(m0 + row) * C_ + k0 + kc]);
            async_copy16(&Bs[bufi][row][kc], &Wt[(size_t)(n0 + row) * C_ + k0 + kc]);
        }
    };

    const int NIT = C_ / 32;
    issue(0, 0);
    for (int it = 0; it < NIT; ++it) {
        wait_async0();           // our tile-it copies have landed
        __syncthreads();         // everyone's landed; everyone done with buf[(it+1)&1]
        if (it + 1 < NIT) issue(32 * (it + 1), (it + 1) & 1);
        const int cb = it & 1;

        Frag a0, a1;
        a0.u[0] = *(const uint4*)&As[cb][wm + lr][kof];
        a0.u[1] = *(const uint4*)&As[cb][wm + lr][16 + kof];
        a1.u[0] = *(const uint4*)&As[cb][wm + 16 + lr][kof];
        a1.u[1] = *(const uint4*)&As[cb][wm + 16 + lr][16 + kof];
#pragma unroll
        for (int j = 0; j < 4; ++j) {
            Frag b;
            b.u[0] = *(const uint4*)&Bs[cb][wn + 16 * j + lr][kof];
            b.u[1] = *(const uint4*)&Bs[cb][wn + 16 * j + lr][16 + kof];
            acc[0][j] = wmma_bf16(a0, b, acc[0][j]);
            acc[1][j] = wmma_bf16(a1, b, acc[1][j]);
        }
    }

    // Epilogue: bias + routed scatter. Block's 128 n-cols sit inside one section.
    const int sect = n0 / C_;          // 0=Q 1=K 2=V
#pragma unroll
    for (int i = 0; i < 2; ++i)
#pragma unroll
        for (int j = 0; j < 4; ++j)
#pragma unroll
            for (int r = 0; r < 8; ++r) {
                int mloc = r + ((lane >> 4) << 3);
                int m = m0 + wm + 16 * i + mloc;
                int n = n0 + wn + 16 * j + lr;
                float v = acc[i][j][r] + bias[n];
                int b = m / T_, t = m - b * T_;
                int nc = n % C_;
                int h = nc / HD, d = nc % HD;
                size_t hb = (size_t)(b * NH + h);
                if (sect == 0)      Q [(hb * T_ + t) * HD + d] = (bf16)v;
                else if (sect == 1) Kx[(hb * T_ + t) * HD + d] = (bf16)v;
                else                Vt[(hb * HD + d) * T_ + t] = (bf16)v;
            }
}

// ---------------- Flash attention: per (b,h), 64 q rows per WG ----------------
// K/V tiles double-buffered via async DMA; one barrier per key block.

__global__ __launch_bounds__(128) void attn(const bf16* __restrict__ Q,
                                            const bf16* __restrict__ Kx,
                                            const bf16* __restrict__ Vt,
                                            bf16* __restrict__ Y) {
    __shared__ bf16 Ks[2][64][72];    // [key][d]
    __shared__ bf16 Vs[2][64][72];    // [d][key]
    __shared__ bf16 Ps[4][16][72];    // per-wave P tile [qrow][key]
    const int tid  = threadIdx.x;
    const int wave = tid >> 5;
    const int lane = tid & 31;
    const int lr   = lane & 15;
    const int kof  = (lane >> 4) * 8;
    const int bh   = blockIdx.x;
    const int b    = bh / NH, h = bh % NH;
    const int q0   = blockIdx.y * 64;
    const int qw   = q0 + 16 * wave;

    const bf16* Qh = Q  + (size_t)bh * T_ * HD;
    const bf16* Kh = Kx + (size_t)bh * T_ * HD;
    const bf16* Vh = Vt + (size_t)bh * HD * T_;

    auto issue = [&](int kb, int bufi) {
        const int kbase = kb * 64;
#pragma unroll
        for (int i = 0; i < 4; ++i) {
            int c   = tid + 128 * i;      // 512 16B chunks per array
            int row = c >> 3;
            int kc  = (c & 7) * 8;
            async_copy16(&Ks[bufi][row][kc], &Kh[(size_t)(kbase + row) * HD + kc]);
            async_copy16(&Vs[bufi][row][kc], &Vh[(size_t)row * T_ + kbase + kc]);
        }
    };

    // Q fragments for this wave's 16 rows (d 0..31 and 32..63)
    Frag qf0, qf1;
    {
        const bf16* qp = &Qh[(size_t)(qw + lr) * HD];
        qf0.u[0] = *(const uint4*)&qp[kof];
        qf0.u[1] = *(const uint4*)&qp[16 + kof];
        qf1.u[0] = *(const uint4*)&qp[32 + kof];
        qf1.u[1] = *(const uint4*)&qp[48 + kof];
    }

    v8f o[4];
#pragma unroll
    for (int j = 0; j < 4; ++j) o[j] = v8f_zero();
    float mrow[8], lsum[8];
#pragma unroll
    for (int r = 0; r < 8; ++r) { mrow[r] = -1e30f; lsum[r] = 0.f; }

    const int nkb = blockIdx.y + 1;   // causal: key blocks 0..qblock
    issue(0, 0);
    for (int kb = 0; kb < nkb; ++kb) {
        wait_async0();
        __syncthreads();
        if (kb + 1 < nkb) issue(kb + 1, (kb + 1) & 1);
        const int cb = kb & 1;
        const int kbase = kb * 64;

        // S = (Q K^T) * scale, masked
        float p[4][8];
        float pm[8];
#pragma unroll
        for (int r = 0; r < 8; ++r) pm[r] = -1e30f;
#pragma unroll
        for (int j = 0; j < 4; ++j) {
            Frag b0, b1;
            b0.u[0] = *(const uint4*)&Ks[cb][16 * j + lr][kof];
            b0.u[1] = *(const uint4*)&Ks[cb][16 * j + lr][16 + kof];
            b1.u[0] = *(const uint4*)&Ks[cb][16 * j + lr][32 + kof];
            b1.u[1] = *(const uint4*)&Ks[cb][16 * j + lr][48 + kof];
            v8f s = v8f_zero();
            s = wmma_bf16(qf0, b0, s);
            s = wmma_bf16(qf1, b1, s);
#pragma unroll
            for (int r = 0; r < 8; ++r) {
                int mloc = r + ((lane >> 4) << 3);
                int qrow = qw + mloc;
                int krow = kbase + 16 * j + lr;
                float sv = s[r] * 0.125f;               // HEAD^-0.5
                sv = (krow <= qrow) ? sv : -1e30f;      // causal mask
                p[j][r] = sv;
                pm[r] = fmaxf(pm[r], sv);
            }
        }
        // row max across the 16 lanes holding each row (wave32-safe)
#pragma unroll
        for (int off = 8; off > 0; off >>= 1)
#pragma unroll
            for (int r = 0; r < 8; ++r)
                pm[r] = fmaxf(pm[r], __shfl_xor(pm[r], off, 32));

        float alpha[8], rs[8];
#pragma unroll
        for (int r = 0; r < 8; ++r) {
            float mn = fmaxf(mrow[r], pm[r]);
            alpha[r] = __expf(mrow[r] - mn);
            mrow[r]  = mn;
            rs[r]    = 0.f;
        }
#pragma unroll
        for (int j = 0; j < 4; ++j)
#pragma unroll
            for (int r = 0; r < 8; ++r) {
                float pv = __expf(p[j][r] - mrow[r]);
                p[j][r] = pv;
                rs[r] += pv;
            }
#pragma unroll
        for (int off = 8; off > 0; off >>= 1)
#pragma unroll
            for (int r = 0; r < 8; ++r)
                rs[r] += __shfl_xor(rs[r], off, 32);
#pragma unroll
        for (int r = 0; r < 8; ++r) lsum[r] = lsum[r] * alpha[r] + rs[r];
#pragma unroll
        for (int j = 0; j < 4; ++j)
#pragma unroll
            for (int r = 0; r < 8; ++r) o[j][r] = o[j][r] * alpha[r];

        // C-layout -> A-layout via per-wave LDS tile (wave-private: no barrier)
#pragma unroll
        for (int j = 0; j < 4; ++j)
#pragma unroll
            for (int r = 0; r < 8; ++r) {
                int mloc = r + ((lane >> 4) << 3);
                Ps[wave][mloc][16 * j + lr] = (bf16)p[j][r];
            }

        // O += P V   (K-dim = 64 keys -> two 32-steps)
#pragma unroll
        for (int st = 0; st < 2; ++st) {
            Frag af;
            af.u[0] = *(const uint4*)&Ps[wave][lr][32 * st + kof];
            af.u[1] = *(const uint4*)&Ps[wave][lr][32 * st + 16 + kof];
#pragma unroll
            for (int j = 0; j < 4; ++j) {
                Frag bf;
                bf.u[0] = *(const uint4*)&Vs[cb][16 * j + lr][32 * st + kof];
                bf.u[1] = *(const uint4*)&Vs[cb][16 * j + lr][32 * st + 16 + kof];
                o[j] = wmma_bf16(af, bf, o[j]);
            }
        }
    }

    // normalize, write y in [B][T][C] bf16 (head h occupies cols h*64..)
#pragma unroll
    for (int j = 0; j < 4; ++j)
#pragma unroll
        for (int r = 0; r < 8; ++r) {
            int mloc = r + ((lane >> 4) << 3);
            int qrow = qw + mloc;
            int d = 16 * j + lr;
            float val = o[j][r] / lsum[r];
            Y[((size_t)(b * T_ + qrow)) * C_ + h * HD + d] = (bf16)val;
        }
}

// ---------------- Output projection: [M_,C_] x [C_,C_] + bias -> f32 ----------------

__global__ __launch_bounds__(256) void proj_gemm(const bf16* __restrict__ Yb,
                                                 const bf16* __restrict__ Wt,
                                                 const float* __restrict__ bias,
                                                 float* __restrict__ out) {
    __shared__ bf16 As[2][128][40];
    __shared__ bf16 Bs[2][128][40];
    const int tid  = threadIdx.x;
    const int wave = tid >> 5;
    const int lane = tid & 31;
    const int lr   = lane & 15;
    const int kof  = (lane >> 4) * 8;
    const int wm   = (wave >> 1) * 32;
    const int wn   = (wave & 1) * 64;
    const int m0   = blockIdx.x * 128;
    const int n0   = blockIdx.y * 128;

    v8f acc[2][4];
#pragma unroll
    for (int i = 0; i < 2; ++i)
#pragma unroll
        for (int j = 0; j < 4; ++j) acc[i][j] = v8f_zero();

    auto issue = [&](int k0, int bufi) {
#pragma unroll
        for (int i = 0; i < 2; ++i) {
            int c   = tid + 256 * i;
            int row = c >> 2;
            int kc  = (c & 3) * 8;
            async_copy16(&As[bufi][row][kc], &Yb[(size_t)(m0 + row) * C_ + k0 + kc]);
            async_copy16(&Bs[bufi][row][kc], &Wt[(size_t)(n0 + row) * C_ + k0 + kc]);
        }
    };

    const int NIT = C_ / 32;
    issue(0, 0);
    for (int it = 0; it < NIT; ++it) {
        wait_async0();
        __syncthreads();
        if (it + 1 < NIT) issue(32 * (it + 1), (it + 1) & 1);
        const int cb = it & 1;

        Frag a0, a1;
        a0.u[0] = *(const uint4*)&As[cb][wm + lr][kof];
        a0.u[1] = *(const uint4*)&As[cb][wm + lr][16 + kof];
        a1.u[0] = *(const uint4*)&As[cb][wm + 16 + lr][kof];
        a1.u[1] = *(const uint4*)&As[cb][wm + 16 + lr][16 + kof];
#pragma unroll
        for (int j = 0; j < 4; ++j) {
            Frag b;
            b.u[0] = *(const uint4*)&Bs[cb][wn + 16 * j + lr][kof];
            b.u[1] = *(const uint4*)&Bs[cb][wn + 16 * j + lr][16 + kof];
            acc[0][j] = wmma_bf16(a0, b, acc[0][j]);
            acc[1][j] = wmma_bf16(a1, b, acc[1][j]);
        }
    }

#pragma unroll
    for (int i = 0; i < 2; ++i)
#pragma unroll
        for (int j = 0; j < 4; ++j)
#pragma unroll
            for (int r = 0; r < 8; ++r) {
                int mloc = r + ((lane >> 4) << 3);
                int m = m0 + wm + 16 * i + mloc;
                int n = n0 + wn + 16 * j + lr;
                out[(size_t)m * C_ + n] = acc[i][j][r] + bias[n];
            }
}

// ---------------- launcher ----------------

extern "C" void kernel_launch(void* const* d_in, const int* in_sizes, int n_in,
                              void* d_out, int out_size, void* d_ws, size_t ws_size,
                              hipStream_t stream) {
    const float* x      = (const float*)d_in[0];
    const float* W_attn = (const float*)d_in[1];
    const float* b_attn = (const float*)d_in[2];
    const float* W_proj = (const float*)d_in[3];
    const float* b_proj = (const float*)d_in[4];
    float* out = (float*)d_out;

    char* p = (char*)d_ws;
    bf16* Xbf  = (bf16*)p; p += (size_t)M_ * C_ * 2;        // 16 MB
    bf16* Wqkv = (bf16*)p; p += (size_t)3 * C_ * C_ * 2;    //  6 MB
    bf16* Wpj  = (bf16*)p; p += (size_t)C_ * C_ * 2;        //  2 MB
    bf16* Qb   = (bf16*)p; p += (size_t)M_ * C_ * 2;        // 16 MB
    bf16* Kb   = (bf16*)p; p += (size_t)M_ * C_ * 2;        // 16 MB
    bf16* Vt   = (bf16*)p; p += (size_t)M_ * C_ * 2;        // 16 MB
    bf16* Yb   = (bf16*)p; p += (size_t)M_ * C_ * 2;        // 16 MB  (total 88 MB)

    const int nx = M_ * C_;
    cast_f32_bf16<<<(nx + 255) / 256, 256, 0, stream>>>(x, Xbf, nx);
    transpose_f32_bf16<<<(C_ * 3 * C_ + 255) / 256, 256, 0, stream>>>(W_attn, Wqkv, C_, 3 * C_);
    transpose_f32_bf16<<<(C_ * C_ + 255) / 256, 256, 0, stream>>>(W_proj, Wpj, C_, C_);

    qkv_gemm<<<dim3(M_ / 128, (3 * C_) / 128), 256, 0, stream>>>(Xbf, Wqkv, b_attn, Qb, Kb, Vt);
    attn<<<dim3(B_ * NH, T_ / 64), 128, 0, stream>>>(Qb, Kb, Vt, Yb);
    proj_gemm<<<dim3(M_ / 128, C_ / 128), 256, 0, stream>>>(Yb, Wpj, b_proj, out);
}
